// Seq2seq_Attention_89172110999865
// MI455X (gfx1250) — matrix operand order repaired and checked by reading, not compiled, not run on hardware
//
#include <hip/hip_runtime.h>
#include <hip/hip_bf16.h>

typedef __attribute__((ext_vector_type(16))) __bf16 bf16x16;
typedef __attribute__((ext_vector_type(8)))  float  f32x8;

#define DEV static __device__ __forceinline__

// ---------- helpers ----------
DEV unsigned short f2bf(float f) {
  union { float f; unsigned u; } c; c.f = f;
  unsigned u = c.u;
  return (unsigned short)((u + 0x7FFFu + ((u >> 16) & 1u)) >> 16); // RNE
}

union TileU { bf16x16 v; uint4 q[2]; };

DEV float waveSum(float x) {
#pragma unroll
  for (int o = 16; o > 0; o >>= 1) x += __shfl_down(x, o, 32);
  return x;
}
DEV float waveMax(float x) {
#pragma unroll
  for (int o = 16; o > 0; o >>= 1) x = fmaxf(x, __shfl_down(x, o, 32));
  return x;
}
DEV float blockSum(float x, float* sh) {           // blockDim.x == 256
  const int lane = threadIdx.x & 31, wv = threadIdx.x >> 5;
  x = waveSum(x);
  if (lane == 0) sh[wv] = x;
  __syncthreads();
  if (wv == 0) {
    float v = (lane < 8) ? sh[lane] : 0.f;
    v = waveSum(v);
    if (lane == 0) sh[8] = v;
  }
  __syncthreads();
  float r = sh[8];
  __syncthreads();
  return r;
}
DEV float blockMax(float x, float* sh) {
  const int lane = threadIdx.x & 31, wv = threadIdx.x >> 5;
  x = waveMax(x);
  if (lane == 0) sh[wv] = x;
  __syncthreads();
  if (wv == 0) {
    float v = (lane < 8) ? sh[lane] : -3.402823466e38f;
    v = waveMax(v);
    if (lane == 0) sh[8] = v;
  }
  __syncthreads();
  float r = sh[8];
  __syncthreads();
  return r;
}

// Tile loaders use contiguous k order (k = 16*half + s), same for A and B.
// A (16x32 bf16): lane-half k-offset +16, 16 consecutive k
DEV bf16x16 load_a(const unsigned short* p) {
  TileU t;
  t.q[0] = *(const uint4*)(p);
  t.q[1] = *(const uint4*)(p + 8);
  return t.v;
}
// B (32x16 bf16, from pre-transposed [N,K]): lane-half k-offset +16, 16 consecutive k
DEV bf16x16 load_b(const unsigned short* p) {
  TileU t;
  t.q[0] = *(const uint4*)(p);
  t.q[1] = *(const uint4*)(p + 8);
  return t.v;
}

// ---------- prep kernels ----------
__global__ void gather_embed(const int* __restrict__ seq, const float* __restrict__ emb,
                             unsigned short* __restrict__ out, int D) {
  const int row = blockIdx.x;
  const int v = seq[row];
  const float* e = emb + (size_t)v * D;
  unsigned short* o = out + (size_t)row * D;
  for (int d = threadIdx.x; d < D; d += blockDim.x) o[d] = f2bf(e[d]);
}

__global__ void transpose_to_bf16(const float* __restrict__ src, unsigned short* __restrict__ dst,
                                  int K, int N) { // src [K,N] f32 -> dst [N,K] bf16
  const int idx = blockIdx.x * blockDim.x + threadIdx.x;
  if (idx >= K * N) return;
  const int k = idx / N, n = idx - k * N;
  dst[(size_t)n * K + k] = f2bf(src[idx]);
}

__global__ void zero_h(float* hf, unsigned short* h0, unsigned short* h1, int n) {
  const int i = blockIdx.x * blockDim.x + threadIdx.x;
  if (i < n) { hf[i] = 0.f; h0[i] = 0; h1[i] = 0; }
}

// ---------- big streaming GEMM (M large): wave computes 16x64, A reused across 4 WMMAs ----
// C[M,N] f32 = A[M,K]bf16 @ B ;  BT is [N,K] bf16. N fixed 1024. Block: 8 waves -> 16M x 512N.
__global__ void __launch_bounds__(256)
gemm_big_x4(const unsigned short* __restrict__ A,
            const unsigned short* __restrict__ BT,
            float* __restrict__ Cf, int K)
{
  constexpr int N = 1024;
  const int lane  = threadIdx.x & 31;
  const int wv    = threadIdx.x >> 5;
  const int tileM = blockIdx.y << 4;
  const int tileN = (blockIdx.x << 9) + (wv << 6);   // 64 cols per wave
  const int half  = lane >> 4;
  const int mrow  = tileM + (lane & 15);
  const int nrow  = tileN + (lane & 15);

  f32x8 acc0 = {}, acc1 = {}, acc2 = {}, acc3 = {};

  const unsigned short* pa  = A  + (size_t)mrow * K + (half << 4);
  const unsigned short* pb0 = BT + (size_t)(nrow     ) * K + (half << 4);
  const unsigned short* pb1 = BT + (size_t)(nrow + 16) * K + (half << 4);
  const unsigned short* pb2 = BT + (size_t)(nrow + 32) * K + (half << 4);
  const unsigned short* pb3 = BT + (size_t)(nrow + 48) * K + (half << 4);
  for (int k0 = 0; k0 < K; k0 += 32) {
    const bf16x16 a  = load_a(pa);
    const bf16x16 b0 = load_b(pb0);
    const bf16x16 b1 = load_b(pb1);
    const bf16x16 b2 = load_b(pb2);
    const bf16x16 b3 = load_b(pb3);
    if (k0 + 32 < K) __builtin_prefetch(pa + 32, 0, 1);
    acc0 = __builtin_amdgcn_wmma_f32_16x16x32_bf16(false, a, false, b0, (short)0, acc0, false, false);
    acc1 = __builtin_amdgcn_wmma_f32_16x16x32_bf16(false, a, false, b1, (short)0, acc1, false, false);
    acc2 = __builtin_amdgcn_wmma_f32_16x16x32_bf16(false, a, false, b2, (short)0, acc2, false, false);
    acc3 = __builtin_amdgcn_wmma_f32_16x16x32_bf16(false, a, false, b3, (short)0, acc3, false, false);
    pa += 32; pb0 += 32; pb1 += 32; pb2 += 32; pb3 += 32;
  }

  const int nb = tileN + (lane & 15);
  const int mbase = tileM + (half << 3);
#pragma unroll
  for (int r = 0; r < 8; ++r) {
    float* row = Cf + (size_t)(mbase + r) * N + nb;
    row[0]  = acc0[r];
    row[16] = acc1[r];
    row[32] = acc2[r];
    row[48] = acc3[r];
  }
}

// ---------- small fused GEMM (M=128 recurrence steps): wave computes one 16x16 tile ----
// C = A@B (+ A2@B2) (+ addend) (+ bias), opt tanh; writes f32 + up to two bf16 copies.
__global__ void __launch_bounds__(256)
gemm_bf16_wmma(const unsigned short* __restrict__ A,
               const unsigned short* __restrict__ BT,
               const unsigned short* __restrict__ A2,
               const unsigned short* __restrict__ BT2,
               const float* __restrict__ addend, long long addStride,
               const float* __restrict__ bias,
               float* __restrict__ Cf, long long cStride,
               unsigned short* __restrict__ Cbf,
               unsigned short* __restrict__ Cbf2, long long cbf2Stride,
               int N, int K, int K2, int act)
{
  const int lane  = threadIdx.x & 31;
  const int wv    = threadIdx.x >> 5;
  const int tileM = blockIdx.y << 4;
  const int tileN = (blockIdx.x << 7) + (wv << 4);
  const int half  = lane >> 4;
  const int mrow  = tileM + (lane & 15);
  const int nrow  = tileN + (lane & 15);

  f32x8 acc = {};

  const unsigned short* pa = A  + (size_t)mrow * K + (half << 4);
  const unsigned short* pb = BT + (size_t)nrow * K + (half << 4);
  for (int k0 = 0; k0 < K; k0 += 32) {
    const bf16x16 a = load_a(pa);
    const bf16x16 b = load_b(pb);
    if (k0 + 32 < K) { __builtin_prefetch(pa + 32, 0, 1); __builtin_prefetch(pb + 32, 0, 1); }
    acc = __builtin_amdgcn_wmma_f32_16x16x32_bf16(false, a, false, b, (short)0, acc, false, false);
    pa += 32; pb += 32;
  }
  if (A2) {
    const unsigned short* pa2 = A2  + (size_t)mrow * K2 + (half << 4);
    const unsigned short* pb2 = BT2 + (size_t)nrow * K2 + (half << 4);
    for (int k0 = 0; k0 < K2; k0 += 32) {
      const bf16x16 a = load_a(pa2);
      const bf16x16 b = load_b(pb2);
      if (k0 + 32 < K2) { __builtin_prefetch(pa2 + 32, 0, 1); __builtin_prefetch(pb2 + 32, 0, 1); }
      acc = __builtin_amdgcn_wmma_f32_16x16x32_bf16(false, a, false, b, (short)0, acc, false, false);
      pa2 += 32; pb2 += 32;
    }
  }

  // Epilogue. C layout: vgpr r -> (M = r + 8*half, N = lane&15)
  const int n = tileN + (lane & 15);
  const int mbase = tileM + (half << 3);
#pragma unroll
  for (int r = 0; r < 8; ++r) {
    const int m = mbase + r;
    float x = acc[r];
    if (addend) x += addend[(size_t)m * addStride + n];
    if (bias)   x += bias[n];
    if (act)    x = tanhf(x);
    if (Cf)     Cf[(size_t)m * cStride + n] = x;
    if (Cbf)    Cbf[(size_t)m * N + n] = f2bf(x);
    if (Cbf2)   Cbf2[(size_t)m * cbf2Stride + n] = f2bf(x);
  }
}

// ---------- fused attention (per decoder step) ----------
// e[s,h] = hidA[b,h] + Epart[b,s,h]
// denom[s] = sum_h exp(e - max);  Aacc[h] = sum_s exp/denom
// w[s] = sum_h Aacc[h]*e[s,h];    ctx[h] = sum_s w[s]*e[s,h]
__global__ void __launch_bounds__(256)
attention_ctx(const float* __restrict__ Epart, const float* __restrict__ hidA,
              unsigned short* __restrict__ ctx_bf)
{
  constexpr int S = 256, H = 1024, T = 256, J = H / T;
  __shared__ float sh[12];
  const int b = blockIdx.x;
  const int tid = threadIdx.x;
  float hv[J], Aacc[J] = {0.f, 0.f, 0.f, 0.f}, Cacc[J] = {0.f, 0.f, 0.f, 0.f};
#pragma unroll
  for (int j = 0; j < J; ++j) hv[j] = hidA[(size_t)b * H + tid + j * T];
  const float* Eb = Epart + (size_t)b * S * H;

  for (int s = 0; s < S; ++s) {
    const float* Er = Eb + (size_t)s * H;
    float e[J], mx = -3.402823466e38f;
#pragma unroll
    for (int j = 0; j < J; ++j) { e[j] = hv[j] + Er[tid + j * T]; mx = fmaxf(mx, e[j]); }
    mx = blockMax(mx, sh);
    float ex[J], sum = 0.f;
#pragma unroll
    for (int j = 0; j < J; ++j) { ex[j] = __expf(e[j] - mx); sum += ex[j]; }
    sum = blockSum(sum, sh);
    const float inv = 1.0f / sum;
#pragma unroll
    for (int j = 0; j < J; ++j) Aacc[j] += ex[j] * inv;
  }
  for (int s = 0; s < S; ++s) {
    const float* Er = Eb + (size_t)s * H;
    float e[J], part = 0.f;
#pragma unroll
    for (int j = 0; j < J; ++j) { e[j] = hv[j] + Er[tid + j * T]; part += Aacc[j] * e[j]; }
    const float w = blockSum(part, sh);
#pragma unroll
    for (int j = 0; j < J; ++j) Cacc[j] += w * e[j];
  }
#pragma unroll
  for (int j = 0; j < J; ++j) ctx_bf[(size_t)b * H + tid + j * T] = f2bf(Cacc[j]);
}

// ---------- host orchestration ----------
extern "C" void kernel_launch(void* const* d_in, const int* in_sizes, int n_in,
                              void* d_out, int out_size, void* d_ws, size_t ws_size,
                              hipStream_t stream)
{
  (void)in_sizes; (void)n_in; (void)out_size; (void)ws_size;
  const int*   src_seq = (const int*)  d_in[0];
  const int*   dec_seq = (const int*)  d_in[1];
  const float* emb     = (const float*)d_in[2];
  const float* enc_Wx  = (const float*)d_in[3];
  const float* enc_Wh  = (const float*)d_in[4];
  const float* enc_b   = (const float*)d_in[5];
  const float* dec_Wx  = (const float*)d_in[6];
  const float* dec_Wh  = (const float*)d_in[7];
  const float* dec_b   = (const float*)d_in[8];
  const float* attn_W  = (const float*)d_in[9];

  constexpr int Bn = 128, S = 256, D = 1024, H = 1024;
  constexpr size_t NT = (size_t)Bn * S; // 32768 token rows

  char* ws = (char*)d_ws; size_t off = 0;
  auto alloc = [&](size_t bytes) -> void* {
    off = (off + 255) & ~(size_t)255;
    void* p = ws + off; off += bytes; return p;
  };
  unsigned short* srcE     = (unsigned short*)alloc(NT * D * 2);
  unsigned short* decE     = (unsigned short*)alloc(NT * D * 2);
  unsigned short* encWxT   = (unsigned short*)alloc((size_t)D * H * 2);
  unsigned short* encWhT   = (unsigned short*)alloc((size_t)H * H * 2);
  unsigned short* decWxdT  = (unsigned short*)alloc((size_t)D * H * 2);
  unsigned short* decWxcT  = (unsigned short*)alloc((size_t)H * H * 2);
  unsigned short* decWhT   = (unsigned short*)alloc((size_t)H * H * 2);
  unsigned short* attnTopT = (unsigned short*)alloc((size_t)H * H * 2);
  unsigned short* attnBotT = (unsigned short*)alloc((size_t)H * H * 2);
  float*          Xenc     = (float*)alloc(NT * H * 4);
  float*          Xdec     = (float*)alloc(NT * H * 4);
  unsigned short* encHs    = (unsigned short*)alloc(NT * H * 2);
  float*          Epart    = (float*)alloc(NT * H * 4);
  float*          hf       = (float*)alloc((size_t)Bn * H * 4);
  unsigned short* hb0      = (unsigned short*)alloc((size_t)Bn * H * 2);
  unsigned short* hb1      = (unsigned short*)alloc((size_t)Bn * H * 2);
  float*          hidA     = (float*)alloc((size_t)Bn * H * 4);
  unsigned short* ctx      = (unsigned short*)alloc((size_t)Bn * H * 2);
  unsigned short* hb[2] = { hb0, hb1 };

  const dim3 blk(256);

  // --- prep: embedding gathers (f32 -> bf16) and weight transposes ---
  gather_embed<<<(unsigned)NT, blk, 0, stream>>>(src_seq, emb, srcE, D);
  gather_embed<<<(unsigned)NT, blk, 0, stream>>>(dec_seq, emb, decE, D);
  const int KN = D * H, tpb = 256, tpg = (KN + tpb - 1) / tpb;
  transpose_to_bf16<<<tpg, tpb, 0, stream>>>(enc_Wx,                 encWxT,   D, H);
  transpose_to_bf16<<<tpg, tpb, 0, stream>>>(enc_Wh,                 encWhT,   H, H);
  transpose_to_bf16<<<tpg, tpb, 0, stream>>>(dec_Wx,                 decWxdT,  D, H);
  transpose_to_bf16<<<tpg, tpb, 0, stream>>>(dec_Wx + (size_t)D * H, decWxcT,  H, H);
  transpose_to_bf16<<<tpg, tpb, 0, stream>>>(dec_Wh,                 decWhT,   H, H);
  transpose_to_bf16<<<tpg, tpb, 0, stream>>>(attn_W,                 attnTopT, H, H);
  transpose_to_bf16<<<tpg, tpb, 0, stream>>>(attn_W + (size_t)H * H, attnBotT, H, H);
  zero_h<<<(Bn * H + 255) / 256, blk, 0, stream>>>(hf, hb[0], hb[1], Bn * H);

  // --- big hoisted GEMMs (A reused 4x per wave): Xenc, Xdec ---
  const dim3 gBig(H / 512, (unsigned)(NT / 16));
  gemm_big_x4<<<gBig, blk, 0, stream>>>(srcE, encWxT, Xenc, D);
  gemm_big_x4<<<gBig, blk, 0, stream>>>(decE, decWxdT, Xdec, D);

  // --- encoder recurrence: h = tanh(Xenc[:,t,:] + h@enc_Wh + b); record enc_hs ---
  const dim3 gStep(H / 128, Bn / 16);
  int cur = 0;
  for (int t = 0; t < S; ++t) {
    const int nxt = cur ^ 1;
    gemm_bf16_wmma<<<gStep, blk, 0, stream>>>(hb[cur], encWhT, nullptr, nullptr,
        Xenc + (size_t)t * H, (long long)S * H, enc_b,
        hf, H, hb[nxt], encHs + (size_t)t * H, (long long)S * H,
        H, H, 0, 1);
    cur = nxt;
  }

  // --- step-invariant attention term: Epart = enc_hs @ attn_W[H:] ---
  gemm_big_x4<<<gBig, blk, 0, stream>>>(encHs, attnBotT, Epart, H);

  // --- decoder recurrence ---
  for (int t = 0; t < S; ++t) {
    const int nxt = cur ^ 1;
    // hidA = h @ attn_W[:H]
    gemm_bf16_wmma<<<gStep, blk, 0, stream>>>(hb[cur], attnTopT, nullptr, nullptr,
        nullptr, 0, nullptr, hidA, H, nullptr, nullptr, 0, H, H, 0, 0);
    // fused softmax / reduction chain -> ctx (bf16)
    attention_ctx<<<Bn, blk, 0, stream>>>(Epart, hidA, ctx);
    // h = tanh(Xdec[:,t,:] + ctx@dec_Wx[D:] + h@dec_Wh + b)
    gemm_bf16_wmma<<<gStep, blk, 0, stream>>>(ctx, decWxcT, hb[cur], decWhT,
        Xdec + (size_t)t * H, (long long)S * H, dec_b,
        hf, H, hb[nxt], nullptr, 0, H, H, H, 1);
    cur = nxt;
  }

  hipMemcpyAsync(d_out, hf, (size_t)Bn * H * sizeof(float),
                 hipMemcpyDeviceToDevice, stream);
}
